// Conv_mHCBlock_80951543595208
// MI455X (gfx1250) — matrix operand order repaired and checked by reading, not compiled, not run on hardware
//
#include <hip/hip_runtime.h>
#include <hip/hip_bf16.h>
#include <math.h>

// ---------------- problem constants ----------------
#define BB 16
#define CC_ 768
#define HH_ 56
#define WW_ 56
#define PP 3136          // 56*56
#define NS 4             // streams
#define DD 192           // channels per stream
#define KDIM 1728        // 192*9
#define RMS_EPS 1.1920929e-07f
#define SK_EPS 1e-6f
#define SK_ITERS 20

typedef __bf16    v16bf __attribute__((ext_vector_type(16)));
typedef float     v8f   __attribute__((ext_vector_type(8)));
typedef unsigned int u32x4 __attribute__((ext_vector_type(4)));

union AV { v16bf v; u32x4 q[2]; };

// ---------------- 1) global average pool ----------------
__global__ __launch_bounds__(256) void gap_kernel(const float* __restrict__ x,
                                                  float* __restrict__ xgap) {
  const int bc = blockIdx.x;            // 0 .. B*C-1
  const float* p = x + (size_t)bc * PP;
  const int t = threadIdx.x;
  float s = 0.f;
  for (int i = t; i < PP; i += 256) s += p[i];
  __shared__ float red[256];
  red[t] = s; __syncthreads();
  for (int k = 128; k > 0; k >>= 1) { if (t < k) red[t] += red[t + k]; __syncthreads(); }
  if (t == 0) xgap[bc] = red[0] * (1.f / (float)PP);
}

// ---------------- 2) gates: RMSNorm + 3 linears + sigmoid + sinkhorn ----------------
__global__ __launch_bounds__(256) void gates_kernel(
    const float* __restrict__ xgap,
    const float* __restrict__ w_pre, const float* __restrict__ w_post,
    const float* __restrict__ w_res,
    const float* __restrict__ b_pre, const float* __restrict__ b_post,
    const float* __restrict__ b_res,
    const float* __restrict__ a_pre, const float* __restrict__ a_post,
    const float* __restrict__ a_res,
    const float* __restrict__ rms_w,
    float* __restrict__ gates) {
  const int b = blockIdx.x;
  const int t = threadIdx.x;
  __shared__ float red[256];
  __shared__ float xn[CC_];
  __shared__ float dots[24];

  float ss = 0.f;
  for (int c = t; c < CC_; c += 256) { float v = xgap[b * CC_ + c]; ss += v * v; }
  red[t] = ss; __syncthreads();
  for (int k = 128; k > 0; k >>= 1) { if (t < k) red[t] += red[t + k]; __syncthreads(); }
  const float rms = rsqrtf(red[0] / (float)CC_ + RMS_EPS);
  __syncthreads();
  for (int c = t; c < CC_; c += 256) xn[c] = xgap[b * CC_ + c] * rms * rms_w[c];
  __syncthreads();

  float acc[24];
  #pragma unroll
  for (int j = 0; j < 24; ++j) acc[j] = 0.f;
  for (int c = t; c < CC_; c += 256) {
    const float v = xn[c];
    #pragma unroll
    for (int j = 0; j < 4; ++j)  acc[j]      += v * w_pre[j * CC_ + c];
    #pragma unroll
    for (int j = 0; j < 4; ++j)  acc[4 + j]  += v * w_post[j * CC_ + c];
    #pragma unroll
    for (int j = 0; j < 16; ++j) acc[8 + j]  += v * w_res[j * CC_ + c];
  }
  for (int j = 0; j < 24; ++j) {
    red[t] = acc[j]; __syncthreads();
    for (int k = 128; k > 0; k >>= 1) { if (t < k) red[t] += red[t + k]; __syncthreads(); }
    if (t == 0) dots[j] = red[0];
    __syncthreads();
  }

  if (t == 0) {
    float Hpre[4], Hpost[4], M[16], lg[16];
    const float ap = a_pre[0], apo = a_post[0], ar = a_res[0];
    #pragma unroll
    for (int j = 0; j < 4; ++j) Hpre[j]  = 1.f / (1.f + expf(-(ap  * dots[j]     + b_pre[j])));
    #pragma unroll
    for (int j = 0; j < 4; ++j) Hpost[j] = 2.f / (1.f + expf(-(apo * dots[4 + j] + b_post[j])));
    #pragma unroll
    for (int j = 0; j < 16; ++j) lg[j] = ar * dots[8 + j] + b_res[j];
    for (int r = 0; r < 4; ++r) {
      float mx = lg[r * 4];
      for (int c = 1; c < 4; ++c) mx = fmaxf(mx, lg[r * 4 + c]);
      for (int c = 0; c < 4; ++c) M[r * 4 + c] = expf(lg[r * 4 + c] - mx);
    }
    for (int it = 0; it < SK_ITERS; ++it) {
      for (int r = 0; r < 4; ++r) {
        float s = M[r * 4] + M[r * 4 + 1] + M[r * 4 + 2] + M[r * 4 + 3] + SK_EPS;
        for (int c = 0; c < 4; ++c) M[r * 4 + c] /= s;
      }
      for (int c = 0; c < 4; ++c) {
        float s = M[c] + M[4 + c] + M[8 + c] + M[12 + c] + SK_EPS;
        for (int r = 0; r < 4; ++r) M[r * 4 + c] /= s;
      }
    }
    float* g = gates + b * 40;
    for (int j = 0; j < 4; ++j)  g[j]      = Hpre[j];
    for (int j = 0; j < 4; ++j)  g[4 + j]  = Hpost[j];
    for (int j = 0; j < 16; ++j) g[8 + j]  = M[j];
    for (int n = 0; n < 4; ++n)
      for (int m = 0; m < 4; ++m) g[24 + n * 4 + m] = M[n * 4 + m] * Hpre[m];
  }
}

// ---------------- 3) weight repack: OIHW f32 -> [O][K=(r,s,c)] bf16, + zero page ----------------
__global__ __launch_bounds__(256) void wt_kernel(const float* __restrict__ conv_w,
                                                 __bf16* __restrict__ Wt,
                                                 __bf16* __restrict__ zpage) {
  const int i = blockIdx.x * 256 + threadIdx.x;       // 192*1728 = 331776 exact
  const int o = i / KDIM;
  const int k = i - o * KDIM;
  const int rs = k / DD;
  const int c = k - rs * DD;
  Wt[i] = (__bf16)conv_w[((size_t)o * DD + c) * 9 + rs];
  if (blockIdx.x == 0) zpage[threadIdx.x] = (__bf16)0.0f;   // 256 halves = 512 B
}

// ---------------- 4) stream mix -> x_streams in HWC bf16 ----------------
// x_streams[img = b*4+n][p][d] = sum_m Hres[n,m]*Hpre[m] * x[b, m*192+d, p]
__global__ __launch_bounds__(256) void mix_kernel(const float* __restrict__ x,
                                                  const float* __restrict__ gates,
                                                  __bf16* __restrict__ xs) {
  const int b  = blockIdx.y;
  const int p0 = blockIdx.x * 16;
  const int t  = threadIdx.x;
  __shared__ float lds[CC_ * 17];                      // padded stride 17 -> 52KB

  // phase 1: coalesced NCHW reads, transpose into LDS
  {
    const int pl  = t & 15;
    const int row = t >> 4;                            // 16 rows
    const float* xb = x + (size_t)b * CC_ * PP + p0 + pl;
    for (int c = row; c < CC_; c += 16)
      lds[c * 17 + pl] = xb[(size_t)c * PP];
  }
  __syncthreads();

  const float* g = gates + b * 40 + 24;                // HresPre[n][m]
  // phase 2: mix and store HWC bf16, contiguous in d
  for (int idx = t; idx < NS * 16 * DD; idx += 256) {
    const int n   = idx / (16 * DD);
    const int rem = idx - n * 16 * DD;
    const int p   = rem / DD;
    const int d   = rem - p * DD;
    float v = g[n * 4 + 0] * lds[(0 * DD + d) * 17 + p]
            + g[n * 4 + 1] * lds[(1 * DD + d) * 17 + p]
            + g[n * 4 + 2] * lds[(2 * DD + d) * 17 + p]
            + g[n * 4 + 3] * lds[(3 * DD + d) * 17 + p];
    xs[((size_t)(b * NS + n) * PP + p0 + p) * DD + d] = (__bf16)v;
  }
}

// ---------------- 5) implicit-GEMM conv via bf16 WMMA + fused epilogue ----------------
// grid: x = 98 (spatial tiles of 32), y = 3 (out-channel blocks of 64), z = 64 (images)
// SAME-padding: out-of-bounds lanes point at a 512B zero page; the page covers the
// full cc*32(+16)+hi*8 <= 192-half range, so ALL loads use unconditional immediate
// offsets -> pure global_load_b128 + v_wmma in the hot loop.
__global__ __launch_bounds__(256) void conv_kernel(const __bf16* __restrict__ Wt,
                                                   const __bf16* __restrict__ xs,
                                                   const __bf16* __restrict__ zpage,
                                                   const float* __restrict__ x,
                                                   const float* __restrict__ gates,
                                                   const float* __restrict__ conv_b,
                                                   float* __restrict__ out) {
  const int img = blockIdx.z;
  const int b = img >> 2, n = img & 3;
  const int m0 = blockIdx.y * 64;
  const int p0 = blockIdx.x * 32;
  const int t = threadIdx.x;
  const int lane = t & 31;
  const int l16  = lane & 15;
  const int hi   = lane >> 4;                // 0/1: K-half selector
  const int wave = t >> 5;
  const int wm = wave >> 1;                  // 0..3
  const int wn = wave & 1;                   // 0..1

  const int rowA = m0 + wm * 16 + l16;       // A-operand row (out channel)
  const int colP = p0 + wn * 16 + l16;       // B-operand / output spatial column
  const int h = colP / WW_, w = colP - (colP / WW_) * WW_;

  const __bf16* Arow = Wt + (size_t)rowA * KDIM;
  const __bf16* Ximg = xs + (size_t)img * PP * DD;

  v8f acc = {};

  #pragma unroll
  for (int rs = 0; rs < 9; ++rs) {
    const int r = rs / 3, s = rs - r * 3;
    const int hh = h + r - 1, ww = w + s - 1;
    const bool valid = (hh >= 0) & (hh < HH_) & (ww >= 0) & (ww < WW_);
    const __bf16* Ap = Arow + rs * DD + hi * 8;
    const __bf16* Bp = (valid ? (Ximg + (size_t)(hh * WW_ + ww) * DD)
                              : zpage) + hi * 8;       // zero page covers all offsets below
    #pragma unroll
    for (int cc = 0; cc < 6; ++cc) {
      AV a, bv;
      // A: lane l16 = row M; element e -> K = rs*192 + cc*32 + (e/8)*16 + hi*8 + e%8
      a.q[0] = *(const u32x4*)(Ap + cc * 32);
      a.q[1] = *(const u32x4*)(Ap + cc * 32 + 16);
      // B: lane l16 = col N; element e -> K = rs*192 + cc*32 + (e/8)*16 + hi*8 + e%8
      bv.q[0] = *(const u32x4*)(Bp + cc * 32);
      bv.q[1] = *(const u32x4*)(Bp + cc * 32 + 16);
      acc = __builtin_amdgcn_wmma_f32_16x16x32_bf16(false, a.v, false, bv.v,
                                                    (short)0, acc, false, false);
    }
  }

  // epilogue: bias, H_post gate, fused residual mix (recomputed from x), store
  const float* g = gates + b * 40;
  const float hpost = g[4 + n];
  const float hr0 = g[8 + n * 4 + 0], hr1 = g[8 + n * 4 + 1];
  const float hr2 = g[8 + n * 4 + 2], hr3 = g[8 + n * 4 + 3];
  const float* xb = x + (size_t)b * CC_ * PP + colP;
  #pragma unroll
  for (int v = 0; v < 8; ++v) {
    const int ko = m0 + wm * 16 + hi * 8 + v;          // C/D layout: M = v + (lane/16)*8
    const float y = (acc[v] + conv_b[ko]) * hpost;
    const float rm = hr0 * xb[(size_t)(0 * DD + ko) * PP]
                   + hr1 * xb[(size_t)(1 * DD + ko) * PP]
                   + hr2 * xb[(size_t)(2 * DD + ko) * PP]
                   + hr3 * xb[(size_t)(3 * DD + ko) * PP];
    out[((size_t)b * CC_ + n * DD + ko) * PP + colP] = rm + y;
  }
}

// ---------------- launcher ----------------
extern "C" void kernel_launch(void* const* d_in, const int* in_sizes, int n_in,
                              void* d_out, int out_size, void* d_ws, size_t ws_size,
                              hipStream_t stream) {
  const float* x      = (const float*)d_in[0];
  const float* w_pre  = (const float*)d_in[1];
  const float* w_post = (const float*)d_in[2];
  const float* w_res  = (const float*)d_in[3];
  const float* b_pre  = (const float*)d_in[4];
  const float* b_post = (const float*)d_in[5];
  const float* b_res  = (const float*)d_in[6];
  const float* a_pre  = (const float*)d_in[7];
  const float* a_post = (const float*)d_in[8];
  const float* a_res  = (const float*)d_in[9];
  const float* rms_w  = (const float*)d_in[10];
  const float* conv_w = (const float*)d_in[11];
  const float* conv_b = (const float*)d_in[12];
  float* out = (float*)d_out;

  char* ws = (char*)d_ws;
  float*  xgap  = (float*)(ws);                       // 12288 f = 49152 B
  float*  gates = (float*)(ws + 49152);               // 16*40 f = 2560 B
  __bf16* Wt    = (__bf16*)(ws + 51712);              // 192*1728*2 = 663552 B
  __bf16* xs    = (__bf16*)(ws + 715264);             // 64*3136*192*2 = 77070336 B
  __bf16* zpage = (__bf16*)(ws + 77785600);           // 512 B zero page

  gap_kernel  <<<BB * CC_, 256, 0, stream>>>(x, xgap);
  gates_kernel<<<BB, 256, 0, stream>>>(xgap, w_pre, w_post, w_res,
                                       b_pre, b_post, b_res,
                                       a_pre, a_post, a_res, rms_w, gates);
  wt_kernel   <<<(DD * KDIM) / 256, 256, 0, stream>>>(conv_w, Wt, zpage);
  mix_kernel  <<<dim3(PP / 16, BB), 256, 0, stream>>>(x, gates, xs);
  conv_kernel <<<dim3(PP / 32, DD / 64, BB * NS), 256, 0, stream>>>(Wt, xs, zpage, x,
                                                                    gates, conv_b, out);
}
